// LipMambaBlock_20822001451125
// MI455X (gfx1250) — matrix operand order repaired and checked by reading, not compiled, not run on hardware
//
#include <hip/hip_runtime.h>
#include <math.h>

// ---------------------------------------------------------------------------
// Types for CDNA5 WMMA (wave32)
// ---------------------------------------------------------------------------
typedef __bf16 v16bf __attribute__((ext_vector_type(16)));
typedef __bf16 v8bf  __attribute__((ext_vector_type(8)));
typedef float  v8f   __attribute__((ext_vector_type(8)));

#define DIM   2048
#define DIM2  4096
#define TOK   4096   // BATCH * SEQ
#define NSTATE 16
#define LN_EPS 1e-5f

// ---------------------------------------------------------------------------
// f32 -> bf16 conversion (for weights)
// ---------------------------------------------------------------------------
__global__ void f32_to_bf16_kernel(const float* __restrict__ in,
                                   __bf16* __restrict__ out, int n) {
    int i = (blockIdx.x * blockDim.x + threadIdx.x) * 4;
    if (i + 3 < n) {
        float4 v = *(const float4*)(in + i);
        out[i + 0] = (__bf16)v.x;
        out[i + 1] = (__bf16)v.y;
        out[i + 2] = (__bf16)v.z;
        out[i + 3] = (__bf16)v.w;
    } else {
        for (int k = i; k < n; ++k) out[k] = (__bf16)in[k];
    }
}

// ---------------------------------------------------------------------------
// LayerNorm over D=2048, one block (256 threads) per token. Emits bf16 xn.
// ---------------------------------------------------------------------------
__global__ void layernorm_kernel(const float* __restrict__ x,
                                 const float* __restrict__ gamma,
                                 const float* __restrict__ beta,
                                 __bf16* __restrict__ xn) {
    __shared__ float rs[256];
    __shared__ float rs2[256];
    const int t   = blockIdx.x;
    const int tid = threadIdx.x;
    const float* xr = x + (size_t)t * DIM;

    // each thread owns 8 contiguous elements
    float4 v0 = *(const float4*)(xr + tid * 8);
    float4 v1 = *(const float4*)(xr + tid * 8 + 4);
    float s  = v0.x + v0.y + v0.z + v0.w + v1.x + v1.y + v1.z + v1.w;
    float s2 = v0.x*v0.x + v0.y*v0.y + v0.z*v0.z + v0.w*v0.w
             + v1.x*v1.x + v1.y*v1.y + v1.z*v1.z + v1.w*v1.w;
    rs[tid] = s; rs2[tid] = s2;
    __syncthreads();
    for (int off = 128; off > 0; off >>= 1) {
        if (tid < off) { rs[tid] += rs[tid + off]; rs2[tid] += rs2[tid + off]; }
        __syncthreads();
    }
    const float mu  = rs[0] * (1.0f / DIM);
    const float var = rs2[0] * (1.0f / DIM) - mu * mu;
    const float rstd = rsqrtf(var + LN_EPS);

    __bf16* orow = xn + (size_t)t * DIM;
    const float* g = gamma + tid * 8;
    const float* b = beta + tid * 8;
    float xv[8] = {v0.x, v0.y, v0.z, v0.w, v1.x, v1.y, v1.z, v1.w};
    #pragma unroll
    for (int k = 0; k < 8; ++k)
        orow[tid * 8 + k] = (__bf16)((xv[k] - mu) * rstd * g[k] + b[k]);
}

// ---------------------------------------------------------------------------
// WMMA fragment loaders (bf16, wave32; contiguous k order k = 16*(lane/16) + element, same in A and B)
// A: 16x32 (MxK).  lanes 0-15: h0..15 = K 0..15 of row M=lane
//                  lanes16-31: h0..15 = K 16..31 of row M=lane-16
// ---------------------------------------------------------------------------
__device__ inline v16bf load_afrag(const __bf16* __restrict__ A, int lda,
                                   int m0, int k, int lane) {
    const int lr = lane & 15;
    const int hi = lane >> 4;            // 0 or 1
    const __bf16* p = A + (size_t)(m0 + lr) * lda + k + hi * 16;
    v8bf lo  = *(const v8bf*)p;          // 16B, aligned
    v8bf hi8 = *(const v8bf*)(p + 8);    // 16B, aligned
    v16bf r;
    #pragma unroll
    for (int i = 0; i < 8; ++i) { r[i] = lo[i]; r[8 + i] = hi8[i]; }
    return r;
}

// B: 32x16 (KxN) with B[k][n] = W[n][k] (W row-major E x K).
// lanes 0-15: h0..15 = K 0..15 of column N=lane; lanes 16-31: K 16..31.
__device__ inline v16bf load_bfrag(const __bf16* __restrict__ W, int ldb,
                                   int n0, int k, int lane) {
    const int lr = lane & 15;
    const int hi = lane >> 4;
    const __bf16* p = W + (size_t)(n0 + lr) * ldb + k + hi * 16;
    return *(const v16bf*)p;             // 32B contiguous, aligned
}

// ---------------------------------------------------------------------------
// Tiled bf16 WMMA GEMM:  out[m][e] = sum_k A[m][k] * W[e][k]  (+ epilogue)
// Block: 256 threads = 8 waves (4 M-groups x 2 N-groups), tile 128(M) x 128(E)
// Wave tile: 32(M) x 64(E) = 2x4 WMMA tiles -> 8 WMMA per K-step / 12 loads.
// MODE 0: + bias, write f32 xz; cols < E/2 also written as bf16 x_ssm
// MODE 1: + bias, softplus, clamp to 1.0, write f32 dt
// MODE 2: + bias + residual, write f32 final output
// ---------------------------------------------------------------------------
template <int MODE>
__global__ void gemm_bf16_kernel(const __bf16* __restrict__ A,
                                 const __bf16* __restrict__ W,
                                 const float* __restrict__ bias,
                                 float* __restrict__ outF,
                                 __bf16* __restrict__ outH,
                                 const float* __restrict__ resid,
                                 int M, int E, int K) {
    const int lane = threadIdx.x & 31;
    const int w    = threadIdx.x >> 5;        // 0..7
    const int m0   = blockIdx.y * 128 + (w & 3) * 32;   // 4 M-groups of 32
    const int n0   = blockIdx.x * 128 + (w >> 2) * 64;  // 2 N-groups of 64

    v8f acc[2][4] = {};

    for (int k = 0; k < K; k += 32) {
        v16bf a0 = load_afrag(A, K, m0,      k, lane);
        v16bf a1 = load_afrag(A, K, m0 + 16, k, lane);
        v16bf b0 = load_bfrag(W, K, n0,      k, lane);
        v16bf b1 = load_bfrag(W, K, n0 + 16, k, lane);
        v16bf b2 = load_bfrag(W, K, n0 + 32, k, lane);
        v16bf b3 = load_bfrag(W, K, n0 + 48, k, lane);
        acc[0][0] = __builtin_amdgcn_wmma_f32_16x16x32_bf16(false, a0, false, b0,
                                                            (short)0, acc[0][0], false, false);
        acc[0][1] = __builtin_amdgcn_wmma_f32_16x16x32_bf16(false, a0, false, b1,
                                                            (short)0, acc[0][1], false, false);
        acc[0][2] = __builtin_amdgcn_wmma_f32_16x16x32_bf16(false, a0, false, b2,
                                                            (short)0, acc[0][2], false, false);
        acc[0][3] = __builtin_amdgcn_wmma_f32_16x16x32_bf16(false, a0, false, b3,
                                                            (short)0, acc[0][3], false, false);
        acc[1][0] = __builtin_amdgcn_wmma_f32_16x16x32_bf16(false, a1, false, b0,
                                                            (short)0, acc[1][0], false, false);
        acc[1][1] = __builtin_amdgcn_wmma_f32_16x16x32_bf16(false, a1, false, b1,
                                                            (short)0, acc[1][1], false, false);
        acc[1][2] = __builtin_amdgcn_wmma_f32_16x16x32_bf16(false, a1, false, b2,
                                                            (short)0, acc[1][2], false, false);
        acc[1][3] = __builtin_amdgcn_wmma_f32_16x16x32_bf16(false, a1, false, b3,
                                                            (short)0, acc[1][3], false, false);
    }

    const int lr = lane & 15;
    const int hi = lane >> 4;

    #pragma unroll
    for (int ti = 0; ti < 2; ++ti) {
        #pragma unroll
        for (int tj = 0; tj < 4; ++tj) {
            const int col = n0 + 16 * tj + lr;
            const float bv = bias[col];
            #pragma unroll
            for (int r = 0; r < 8; ++r) {
                const int row = m0 + 16 * ti + hi * 8 + r;  // C/D VGPR layout
                float v = acc[ti][tj][r] + bv;
                const size_t idx = (size_t)row * E + col;
                if constexpr (MODE == 0) {
                    outF[idx] = v;
                    if (col < (E >> 1))
                        outH[(size_t)row * (E >> 1) + col] = (__bf16)v;
                } else if constexpr (MODE == 1) {
                    float sp = (v > 20.0f) ? v : log1pf(__expf(v));
                    outF[idx] = fminf(sp, 1.0f);
                } else {
                    outF[idx] = v + resid[idx];
                }
            }
        }
    }
}

// ---------------------------------------------------------------------------
// B/C projections (N=16 each): one wave32 per token.
// Lanes 0-15 compute B_n, lanes 16-31 compute C_n; shfl(lane^16) forms C*B.
// x_ssm is read from the f32 xz buffer (row stride DIM2, cols [0,DIM)).
// ---------------------------------------------------------------------------
__global__ void bc_kernel(const float* __restrict__ xz,
                          const float* __restrict__ W_B,
                          const float* __restrict__ b_B,
                          const float* __restrict__ W_C,
                          const float* __restrict__ b_C,
                          float* __restrict__ CB) {
    const int wave = threadIdx.x >> 5;
    const int lane = threadIdx.x & 31;
    const int t    = blockIdx.x * 8 + wave;
    const int n    = lane & 15;
    const float* xr = xz + (size_t)t * DIM2;         // x_ssm half
    const float* wr = ((lane < 16) ? W_B : W_C) + (size_t)n * DIM;
    const float  bv = ((lane < 16) ? b_B : b_C)[n];

    float acc = 0.f;
    for (int d = 0; d < DIM; d += 4) {
        float4 xv = *(const float4*)(xr + d);        // broadcast across lanes
        float4 wv = *(const float4*)(wr + d);
        acc = fmaf(xv.x, wv.x, acc);
        acc = fmaf(xv.y, wv.y, acc);
        acc = fmaf(xv.z, wv.z, acc);
        acc = fmaf(xv.w, wv.w, acc);
    }
    acc += bv;
    float other = __shfl(acc, lane ^ 16, 32);        // pair B_n with C_n
    if (lane < 16) CB[(size_t)t * NSTATE + n] = acc * other;
}

// ---------------------------------------------------------------------------
// SSM elementwise:  gated = (sum_n CB_n * exp(A[n,d]*dt)) * x_ssm * silu(z)
// One block of 256 threads covers 256 d's of one token; CB via LDS broadcast.
// ---------------------------------------------------------------------------
__global__ void ssm_kernel(const float* __restrict__ xz,
                           const float* __restrict__ dt,
                           const float* __restrict__ CB,
                           const float* __restrict__ state_A,
                           __bf16* __restrict__ gated) {
    __shared__ float sCB[NSTATE];
    const int t = blockIdx.y;
    const int d = blockIdx.x * 256 + threadIdx.x;
    if (threadIdx.x < NSTATE) sCB[threadIdx.x] = CB[(size_t)t * NSTATE + threadIdx.x];
    __syncthreads();

    const float xs = xz[(size_t)t * DIM2 + d];
    const float zv = xz[(size_t)t * DIM2 + DIM + d];
    const float dv = dt[(size_t)t * DIM + d];
    float acc = 0.f;
    #pragma unroll
    for (int n = 0; n < NSTATE; ++n)
        acc = fmaf(sCB[n], __expf(state_A[n * DIM + d] * dv), acc);

    const float y   = acc * xs;
    const float sig = 1.0f / (1.0f + __expf(-zv));
    gated[(size_t)t * DIM + d] = (__bf16)(y * zv * sig);
}

// ---------------------------------------------------------------------------
// Launch
// ---------------------------------------------------------------------------
extern "C" void kernel_launch(void* const* d_in, const int* in_sizes, int n_in,
                              void* d_out, int out_size, void* d_ws, size_t ws_size,
                              hipStream_t stream) {
    const float* x     = (const float*)d_in[0];
    const float* gamma = (const float*)d_in[1];
    const float* beta  = (const float*)d_in[2];
    const float* W_in  = (const float*)d_in[3];
    const float* b_in  = (const float*)d_in[4];
    const float* A_st  = (const float*)d_in[5];
    const float* W_B   = (const float*)d_in[6];
    const float* b_B   = (const float*)d_in[7];
    const float* W_C   = (const float*)d_in[8];
    const float* b_C   = (const float*)d_in[9];
    const float* W_dt  = (const float*)d_in[10];
    const float* b_dt  = (const float*)d_in[11];
    const float* W_out = (const float*)d_in[12];
    const float* b_out = (const float*)d_in[13];
    float* out = (float*)d_out;

    // workspace carve-out (256B aligned)
    char* ws = (char*)d_ws;
    size_t off = 0;
    auto alloc = [&](size_t bytes) -> char* {
        char* p = ws + off;
        off = (off + bytes + 255) & ~(size_t)255;
        return p;
    };
    __bf16* xn_bf   = (__bf16*)alloc((size_t)TOK * DIM * 2);
    __bf16* Win_bf  = (__bf16*)alloc((size_t)DIM2 * DIM * 2);
    __bf16* Wdt_bf  = (__bf16*)alloc((size_t)DIM * DIM * 2);
    __bf16* Wout_bf = (__bf16*)alloc((size_t)DIM * DIM * 2);
    float*  xz_f    = (float*) alloc((size_t)TOK * DIM2 * 4);
    __bf16* xssm_bf = (__bf16*)alloc((size_t)TOK * DIM * 2);
    float*  dt_f    = (float*) alloc((size_t)TOK * DIM * 4);
    float*  CB_f    = (float*) alloc((size_t)TOK * NSTATE * 4);
    __bf16* gated_bf= (__bf16*)alloc((size_t)TOK * DIM * 2);

    // 1. weight conversion f32 -> bf16
    {
        int n = DIM2 * DIM;
        f32_to_bf16_kernel<<<(n / 4 + 255) / 256, 256, 0, stream>>>(W_in, Win_bf, n);
        n = DIM * DIM;
        f32_to_bf16_kernel<<<(n / 4 + 255) / 256, 256, 0, stream>>>(W_dt, Wdt_bf, n);
        f32_to_bf16_kernel<<<(n / 4 + 255) / 256, 256, 0, stream>>>(W_out, Wout_bf, n);
    }

    // 2. LayerNorm -> bf16 xn
    layernorm_kernel<<<TOK, 256, 0, stream>>>(x, gamma, beta, xn_bf);

    // 3. xz = xn @ W_in^T + b_in  (f32 xz + bf16 x_ssm)
    gemm_bf16_kernel<0><<<dim3(DIM2 / 128, TOK / 128), 256, 0, stream>>>(
        xn_bf, Win_bf, b_in, xz_f, xssm_bf, nullptr, TOK, DIM2, DIM);

    // 4. dt = clamp(softplus(x_ssm @ W_dt^T + b_dt), 1.0)
    gemm_bf16_kernel<1><<<dim3(DIM / 128, TOK / 128), 256, 0, stream>>>(
        xssm_bf, Wdt_bf, b_dt, dt_f, nullptr, nullptr, TOK, DIM, DIM);

    // 5. CB[t,n] = (x_ssm @ W_B^T + b_B)_n * (x_ssm @ W_C^T + b_C)_n
    bc_kernel<<<TOK / 8, 256, 0, stream>>>(xz_f, W_B, b_B, W_C, b_C, CB_f);

    // 6. gated = (sum_n CB_n exp(A*dt)) * x_ssm * silu(z)  -> bf16
    ssm_kernel<<<dim3(DIM / 256, TOK), 256, 0, stream>>>(xz_f, dt_f, CB_f, A_st, gated_bf);

    // 7. out = gated @ W_out^T + b_out + residual
    gemm_bf16_kernel<2><<<dim3(DIM / 128, TOK / 128), 256, 0, stream>>>(
        gated_bf, Wout_bf, b_out, out, nullptr, x, TOK, DIM, DIM);
}
